// Qwen3MoeSparseMoeBlock_62311385530820
// MI455X (gfx1250) — matrix operand order repaired and checked by reading, not compiled, not run on hardware
//
#include <hip/hip_runtime.h>
#include <hip/hip_bf16.h>
#include <cstdint>
#include <cstddef>

// ---------------- problem constants ----------------
#define TOK 4096            // B*S tokens
#define HDIM 2048
#define NEXP 8
#define IDIM 1408
#define NSLOT 8192          // TOK * TOPK
#define MTILE 128
#define NTILE 64
#define SLOT_PAD (NSLOT + MTILE)
#define OUT_ELEMS (TOK * HDIM)
#define MAX_TILES 80        // worst case: 8192/128 + 7 boundary tiles = 71

typedef __bf16 bf16_t;
typedef __attribute__((ext_vector_type(16))) __bf16 v16bf;
typedef __attribute__((ext_vector_type(8)))  float  v8f;

union FragB { uint4 q[2]; v16bf v; };

__device__ __forceinline__ unsigned short f2bf(float f) {
  unsigned u = __float_as_uint(f);
  u += 0x7fffu + ((u >> 16) & 1u);        // round-to-nearest-even
  return (unsigned short)(u >> 16);
}
__device__ __forceinline__ unsigned packbf(float lo, float hi) {
  return (unsigned)f2bf(lo) | ((unsigned)f2bf(hi) << 16);
}

// async 16B global -> LDS copy, x2 (second one at +16B on both sides).
// ldsaddr = low 32 bits of generic LDS pointer (= LDS byte offset);
// tracked by ASYNCcnt, waited with s_wait_asynccnt before the tile barrier.
__device__ __forceinline__ void async_cp32(unsigned ldsaddr, const void* gsrc) {
  asm volatile(
      "global_load_async_to_lds_b128 %0, %1, off\n\t"
      "global_load_async_to_lds_b128 %0, %1, off offset:16"
      :: "v"(ldsaddr), "v"((unsigned long long)(uintptr_t)gsrc)
      : "memory");
}
__device__ __forceinline__ void async_wait0() {
  asm volatile("s_wait_asynccnt 0x0" ::: "memory");
}

// ---------------- workspace layout (bytes) ----------------
// meta ints: [0..7]=counts, [8..16]=offsets, [17..24]=cursor, [25]=ntiles
#define WSO_TILES   1024
#define WSO_TOPI    5120
#define WSO_TOPW    (WSO_TOPI + TOK*2*4)
#define WSO_STOK    (WSO_TOPW + TOK*2*4)
#define WSO_SW      (WSO_STOK + SLOT_PAD*4)
#define WSO_XBF     (WSO_SW   + SLOT_PAD*4)
#define WSO_HBUF    (WSO_XBF  + (size_t)TOK*HDIM*2)

// ---------------- kernel 0: zero output + meta ----------------
__global__ void k_init(float* __restrict__ out, int* __restrict__ meta) {
  size_t idx = ((size_t)blockIdx.x * 256 + threadIdx.x) * 16;
  float4 z = make_float4(0.f, 0.f, 0.f, 0.f);
  float4* p = (float4*)(out + idx);
  p[0] = z; p[1] = z; p[2] = z; p[3] = z;
  if (blockIdx.x == 0 && threadIdx.x < 32) meta[threadIdx.x] = 0;
}

// ---------------- kernel 1: router (one wave per token) ----------------
__global__ __launch_bounds__(256) void k_router(
    const float* __restrict__ x, const float* __restrict__ Wr,
    float* __restrict__ logits, int* __restrict__ top_i,
    float* __restrict__ top_w, int* __restrict__ counts) {
  int wave = threadIdx.x >> 5, lane = threadIdx.x & 31;
  int t = blockIdx.x * 8 + wave;
  float a[8];
#pragma unroll
  for (int e = 0; e < 8; ++e) a[e] = 0.f;
  for (int j = 0; j < HDIM / 32; ++j) {
    int k = j * 32 + lane;
    float xv = x[(size_t)t * HDIM + k];
    const float4* wr = (const float4*)(Wr + (size_t)k * 8);
    float4 w0 = wr[0], w1 = wr[1];
    a[0] += xv * w0.x; a[1] += xv * w0.y; a[2] += xv * w0.z; a[3] += xv * w0.w;
    a[4] += xv * w1.x; a[5] += xv * w1.y; a[6] += xv * w1.z; a[7] += xv * w1.w;
  }
#pragma unroll
  for (int off = 16; off > 0; off >>= 1)
#pragma unroll
    for (int e = 0; e < 8; ++e) a[e] += __shfl_xor(a[e], off, 32);
  if (lane == 0) {
#pragma unroll
    for (int e = 0; e < 8; ++e) logits[(size_t)t * 8 + e] = a[e];
    int i1 = 0;
    for (int e = 1; e < 8; ++e) if (a[e] > a[i1]) i1 = e;   // ties -> lowest idx
    int i2 = (i1 == 0) ? 1 : 0;
    for (int e = 0; e < 8; ++e) if (e != i1 && a[e] > a[i2]) i2 = e;
    // normalized top-2 softmax weights: denominator cancels
    float e2 = __expf(a[i2] - a[i1]);
    float rs = __builtin_amdgcn_rcpf(1.f + e2);
    top_i[t * 2] = i1;      top_i[t * 2 + 1] = i2;
    top_w[t * 2] = rs;      top_w[t * 2 + 1] = e2 * rs;
    atomicAdd(&counts[i1], 1);
    atomicAdd(&counts[i2], 1);
  }
}

// ---------------- kernel 2: offsets + tile descriptor list ----------------
__global__ void k_offsets(int* __restrict__ meta, int* __restrict__ tiles) {
  if (threadIdx.x != 0 || blockIdx.x != 0) return;
  int* counts = meta;
  int* off    = meta + 8;
  int* cur    = meta + 17;
  off[0] = 0;
  for (int e = 0; e < NEXP; ++e) off[e + 1] = off[e] + counts[e];
  for (int e = 0; e < NEXP; ++e) cur[e] = off[e];
  int nt = 0;
  for (int e = 0; e < NEXP; ++e)
    for (int r0 = 0; r0 < counts[e]; r0 += MTILE) {
      int c = counts[e] - r0; if (c > MTILE) c = MTILE;
      tiles[nt * 4 + 0] = e;
      tiles[nt * 4 + 1] = off[e] + r0;
      tiles[nt * 4 + 2] = c;
      ++nt;
    }
  meta[25] = nt;
}

// ---------------- kernel 3: scatter tokens into expert-sorted slots ----------------
__global__ void k_scatter(const int* __restrict__ top_i, const float* __restrict__ top_w,
                          int* __restrict__ meta, int* __restrict__ slot_token,
                          float* __restrict__ slot_w) {
  int t = blockIdx.x * 256 + threadIdx.x;
  if (t >= TOK) return;
  int* cur = meta + 17;
#pragma unroll
  for (int k = 0; k < 2; ++k) {
    int e = top_i[t * 2 + k];
    int pos = atomicAdd(&cur[e], 1);
    slot_token[pos] = t;
    slot_w[pos] = top_w[t * 2 + k];
  }
}

// ---------------- kernel 4: x f32 -> bf16 (done once, reused by both GEMMs) ----------------
__global__ void k_convert(const float* __restrict__ x, unsigned short* __restrict__ Xbf) {
  size_t idx = ((size_t)blockIdx.x * 256 + threadIdx.x) * 8;
  float4 a = *(const float4*)(x + idx);
  float4 b = *(const float4*)(x + idx + 4);
  uint4 o;
  o.x = packbf(a.x, a.y);
  o.y = packbf(a.z, a.w);
  o.z = packbf(b.x, b.y);
  o.w = packbf(b.z, b.w);
  *(uint4*)(Xbf + idx) = o;
}

// ---------------- kernel 5: gate+up GEMM + SiLU*up -> Hbuf (bf16) ----------------
// block: 256 thr (8 waves). tile: M=128 slots x N=64 (I), K=HDIM step 32, LDS double-buffered.
// A tile staged with GLOBAL_LOAD_ASYNC_TO_LDS_B128 (ASYNCcnt); B converted f32->bf16, packed.
// wave grid 4(M) x 2(N); each wave: 2x2 fragments per matrix -> 8 wmma / k-step.
__global__ __launch_bounds__(256) void k_gateup(
    const unsigned short* __restrict__ Xbf, const float* __restrict__ Wg,
    const float* __restrict__ Wu, const int* __restrict__ slot_token,
    const int* __restrict__ tiles, const int* __restrict__ meta,
    unsigned short* __restrict__ Hbuf) {
  int bt = blockIdx.x;
  if (bt >= meta[25]) return;
  int e = tiles[bt * 4], row0 = tiles[bt * 4 + 1], cnt = tiles[bt * 4 + 2];
  int i0 = blockIdx.y * NTILE;

  __shared__ unsigned short Ab[2][MTILE * 32];   // A tile, row-major [m][k]
  __shared__ unsigned short Bg[2][NTILE * 32];   // Wg tile, transposed [n][k]
  __shared__ unsigned short Bu[2][NTILE * 32];   // Wu tile, transposed [n][k]
  __shared__ int tok[MTILE];

  int tid = threadIdx.x;
  if (tid < MTILE) tok[tid] = (tid < cnt) ? slot_token[row0 + tid] : 0;
  __syncthreads();

  int lane = tid & 31, wv = tid >> 5;
  int wm = wv & 3, wn = wv >> 2;
  int r = lane & 15, hi = lane >> 4;

  // per-thread loop-invariant staging state
  int m_a = tid >> 1, p_a = tid & 1;
  const unsigned short* asrc = Xbf + (size_t)tok[m_a] * HDIM + p_a * 16;  // masked rows -> token 0
  unsigned alds[2] = {
    (unsigned)(uintptr_t)(Ab[0] + m_a * 32 + p_a * 16),
    (unsigned)(uintptr_t)(Ab[1] + m_a * 32 + p_a * 16)
  };
  int kb2 = (tid >> 4) << 1;     // 0,2,...,30
  int nb  = (tid & 15) << 2;     // 0,4,...,60
  int kh  = kb2 >> 1;
  const size_t wbase = (size_t)e * HDIM * IDIM + i0;
  const float* gsrc = Wg + wbase + (size_t)kb2 * IDIM + nb;
  const float* usrc = Wu + wbase + (size_t)kb2 * IDIM + nb;

  v8f accG[2][2] = {}; v8f accU[2][2] = {};

  auto stage = [&](int b, int kk) {
    async_cp32(alds[b], asrc + kk);                       // A: 32B/thread, async -> LDS
    const float* gs = gsrc + (size_t)kk * IDIM;
    const float* us = usrc + (size_t)kk * IDIM;
    float4 g0 = *(const float4*)gs, g1 = *(const float4*)(gs + IDIM);
    float4 u0 = *(const float4*)us, u1 = *(const float4*)(us + IDIM);
    unsigned* gw = (unsigned*)Bg[b];
    unsigned* uw = (unsigned*)Bu[b];
    gw[(nb + 0) * 16 + kh] = packbf(g0.x, g1.x);
    gw[(nb + 1) * 16 + kh] = packbf(g0.y, g1.y);
    gw[(nb + 2) * 16 + kh] = packbf(g0.z, g1.z);
    gw[(nb + 3) * 16 + kh] = packbf(g0.w, g1.w);
    uw[(nb + 0) * 16 + kh] = packbf(u0.x, u1.x);
    uw[(nb + 1) * 16 + kh] = packbf(u0.y, u1.y);
    uw[(nb + 2) * 16 + kh] = packbf(u0.z, u1.z);
    uw[(nb + 3) * 16 + kh] = packbf(u0.w, u1.w);
  };

  auto computeStep = [&](int b) {
    FragB a[2], bg[2], bu[2];
    const uint4* A4 = (const uint4*)Ab[b];
#pragma unroll
    for (int mf = 0; mf < 2; ++mf) {            // A: lane%16=row, K {0-7,16-23}/{8-15,24-31}
      int m = wm * 32 + mf * 16 + r;
      a[mf].q[0] = A4[m * 4 + hi];
      a[mf].q[1] = A4[m * 4 + 2 + hi];
    }
    const uint4* G4 = (const uint4*)Bg[b];
    const uint4* U4 = (const uint4*)Bu[b];
#pragma unroll
    for (int nf = 0; nf < 2; ++nf) {            // B: lane%16=col, K {0-7,16-23}/{8-15,24-31}
      int n = wn * 32 + nf * 16 + r;
      bg[nf].q[0] = G4[n * 4 + hi]; bg[nf].q[1] = G4[n * 4 + 2 + hi];
      bu[nf].q[0] = U4[n * 4 + hi]; bu[nf].q[1] = U4[n * 4 + 2 + hi];
    }
#pragma unroll
    for (int mf = 0; mf < 2; ++mf)
#pragma unroll
      for (int nf = 0; nf < 2; ++nf) {
        accG[mf][nf] = __builtin_amdgcn_wmma_f32_16x16x32_bf16(
            false, a[mf].v, false, bg[nf].v, (short)0, accG[mf][nf], false, false);
        accU[mf][nf] = __builtin_amdgcn_wmma_f32_16x16x32_bf16(
            false, a[mf].v, false, bu[nf].v, (short)0, accU[mf][nf], false, false);
      }
  };

  stage(0, 0);
  int buf = 0;
  for (int kk = 0; kk < HDIM; kk += 32) {
    async_wait0();                 // async A-tile of 'buf' complete
    __syncthreads();               // + B-tile visible block-wide
    if (kk + 32 < HDIM) stage(buf ^ 1, kk + 32);
    computeStep(buf);
    buf ^= 1;
  }

  // epilogue: h = silu(g) * u  -> bf16 Hbuf[slot][i]
#pragma unroll
  for (int mf = 0; mf < 2; ++mf)
#pragma unroll
    for (int nf = 0; nf < 2; ++nf)
#pragma unroll
      for (int v = 0; v < 8; ++v) {
        int m = wm * 32 + mf * 16 + hi * 8 + v;   // C: VGPR v -> row v / v+8
        if (m < cnt) {
          float g = accG[mf][nf][v], u = accU[mf][nf][v];
          float hval = g * __builtin_amdgcn_rcpf(1.f + __expf(-g)) * u;
          int ii = i0 + wn * 32 + nf * 16 + r;
          Hbuf[(size_t)(row0 + m) * IDIM + ii] = f2bf(hval);
        }
      }
}

// ---------------- kernel 6: down GEMM, scale by token weight, atomic-add into out ----------------
__global__ __launch_bounds__(256) void k_down(
    const unsigned short* __restrict__ Hbuf, const float* __restrict__ Wd,
    const int* __restrict__ slot_token, const float* __restrict__ slot_w,
    const int* __restrict__ tiles, const int* __restrict__ meta,
    float* __restrict__ out) {
  int bt = blockIdx.x;
  if (bt >= meta[25]) return;
  int e = tiles[bt * 4], row0 = tiles[bt * 4 + 1], cnt = tiles[bt * 4 + 2];
  int n0 = blockIdx.y * NTILE;

  __shared__ unsigned short Ab[2][MTILE * 32];
  __shared__ unsigned short Bd[2][NTILE * 32];
  __shared__ int tok[MTILE];
  __shared__ float sw[MTILE];

  int tid = threadIdx.x;
  if (tid < MTILE) {
    bool ok = tid < cnt;
    tok[tid] = ok ? slot_token[row0 + tid] : 0;
    sw[tid]  = ok ? slot_w[row0 + tid] : 0.f;
  }
  __syncthreads();

  int lane = tid & 31, wv = tid >> 5;
  int wm = wv & 3, wn = wv >> 2;
  int r = lane & 15, hi = lane >> 4;

  int m_a = tid >> 1, p_a = tid & 1;
  const unsigned short* asrc = Hbuf + (size_t)(row0 + m_a) * IDIM + p_a * 16;
  unsigned alds[2] = {
    (unsigned)(uintptr_t)(Ab[0] + m_a * 32 + p_a * 16),
    (unsigned)(uintptr_t)(Ab[1] + m_a * 32 + p_a * 16)
  };
  int kb2 = (tid >> 4) << 1;
  int nb  = (tid & 15) << 2;
  int kh  = kb2 >> 1;
  const size_t wbase = (size_t)e * IDIM * HDIM + n0;
  const float* dsrc = Wd + wbase + (size_t)kb2 * HDIM + nb;

  v8f acc[2][2] = {};

  auto stage = [&](int b, int kk) {
    async_cp32(alds[b], asrc + kk);                       // A from Hbuf (bf16, contiguous slots)
    const float* ds = dsrc + (size_t)kk * HDIM;
    float4 d0 = *(const float4*)ds, d1 = *(const float4*)(ds + HDIM);
    unsigned* dw = (unsigned*)Bd[b];
    dw[(nb + 0) * 16 + kh] = packbf(d0.x, d1.x);
    dw[(nb + 1) * 16 + kh] = packbf(d0.y, d1.y);
    dw[(nb + 2) * 16 + kh] = packbf(d0.z, d1.z);
    dw[(nb + 3) * 16 + kh] = packbf(d0.w, d1.w);
  };

  auto computeStep = [&](int b) {
    FragB a[2], bb[2];
    const uint4* A4 = (const uint4*)Ab[b];
#pragma unroll
    for (int mf = 0; mf < 2; ++mf) {
      int m = wm * 32 + mf * 16 + r;
      a[mf].q[0] = A4[m * 4 + hi];
      a[mf].q[1] = A4[m * 4 + 2 + hi];
    }
    const uint4* B4 = (const uint4*)Bd[b];
#pragma unroll
    for (int nf = 0; nf < 2; ++nf) {
      int n = wn * 32 + nf * 16 + r;
      bb[nf].q[0] = B4[n * 4 + hi]; bb[nf].q[1] = B4[n * 4 + 2 + hi];
    }
#pragma unroll
    for (int mf = 0; mf < 2; ++mf)
#pragma unroll
      for (int nf = 0; nf < 2; ++nf)
        acc[mf][nf] = __builtin_amdgcn_wmma_f32_16x16x32_bf16(
            false, a[mf].v, false, bb[nf].v, (short)0, acc[mf][nf], false, false);
  };

  stage(0, 0);
  int buf = 0;
  for (int kk = 0; kk < IDIM; kk += 32) {
    async_wait0();
    __syncthreads();
    if (kk + 32 < IDIM) stage(buf ^ 1, kk + 32);
    computeStep(buf);
    buf ^= 1;
  }

  // epilogue: out[token][n] += w * y   (2 commutative adds per element -> deterministic)
#pragma unroll
  for (int mf = 0; mf < 2; ++mf)
#pragma unroll
    for (int nf = 0; nf < 2; ++nf)
#pragma unroll
      for (int v = 0; v < 8; ++v) {
        int m = wm * 32 + mf * 16 + hi * 8 + v;
        if (m < cnt) {
          float val = acc[mf][nf][v] * sw[m];
          size_t oi = (size_t)tok[m] * HDIM + n0 + wn * 32 + nf * 16 + r;
          unsafeAtomicAdd(out + oi, val);   // global_atomic_add_f32
        }
      }
}

// ---------------- launcher ----------------
extern "C" void kernel_launch(void* const* d_in, const int* in_sizes, int n_in,
                              void* d_out, int out_size, void* d_ws, size_t ws_size,
                              hipStream_t stream) {
  const float* x  = (const float*)d_in[0];   // [T, H]
  const float* Wr = (const float*)d_in[1];   // [H, E]
  const float* Wg = (const float*)d_in[2];   // [E, H, I]
  const float* Wu = (const float*)d_in[3];   // [E, H, I]
  const float* Wd = (const float*)d_in[4];   // [E, I, H]
  float* out    = (float*)d_out;             // [T, H] then router_logits [T, E]
  float* logits = out + OUT_ELEMS;

  char* ws = (char*)d_ws;
  int*   meta       = (int*)ws;
  int*   tiles      = (int*)(ws + WSO_TILES);
  int*   top_i      = (int*)(ws + WSO_TOPI);
  float* top_w      = (float*)(ws + WSO_TOPW);
  int*   slot_token = (int*)(ws + WSO_STOK);
  float* slot_w     = (float*)(ws + WSO_SW);
  unsigned short* Xbf  = (unsigned short*)(ws + WSO_XBF);
  unsigned short* Hbuf = (unsigned short*)(ws + WSO_HBUF);

  k_init   <<<OUT_ELEMS / (256 * 16), 256, 0, stream>>>(out, meta);
  k_router <<<TOK / 8, 256, 0, stream>>>(x, Wr, logits, top_i, top_w, meta);
  k_offsets<<<1, 32, 0, stream>>>(meta, tiles);
  k_scatter<<<TOK / 256, 256, 0, stream>>>(top_i, top_w, meta, slot_token, slot_w);
  k_convert<<<(TOK * HDIM) / (256 * 8), 256, 0, stream>>>(x, Xbf);
  k_gateup <<<dim3(MAX_TILES, IDIM / NTILE), 256, 0, stream>>>(Xbf, Wg, Wu, slot_token,
                                                               tiles, meta, Hbuf);
  k_down   <<<dim3(MAX_TILES, HDIM / NTILE), 256, 0, stream>>>(Hbuf, Wd, slot_token, slot_w,
                                                               tiles, meta, out);
}
